// ParallelDense_50697793962231
// MI455X (gfx1250) — matrix operand order repaired and checked by reading, not compiled, not run on hardware
//
#include <hip/hip_runtime.h>

// ---- problem constants (match reference) ----
#define B_    4
#define S_    2048
#define P_    8
#define FIN   1024
#define FOUT  1024
#define ROWSTRIDE (P_*FIN)    // 8192 floats between consecutive (b,s) rows
#define OUTSTRIDE (P_*FOUT)   // 8192 floats between consecutive output rows

#define MTILE 128
#define NTILE 128
#define KTILE 32
#define KSTEPS (FIN/KTILE)    // 32
#define LDS_STRIDE 40         // 32 bf16 + 16B pad -> conflict-free frag reads

typedef __bf16 bf16_t;
typedef __bf16 v16bf __attribute__((ext_vector_type(16)));
typedef __bf16 v8bf  __attribute__((ext_vector_type(8)));
typedef __bf16 v4bf  __attribute__((ext_vector_type(4)));
typedef float  v8f   __attribute__((ext_vector_type(8)));
typedef float  v4f   __attribute__((ext_vector_type(4), aligned(16)));

union FragAB { v16bf v; v8bf h[2]; };

__global__ __launch_bounds__(256)
void pdense_bf16_wmma(const float* __restrict__ X,
                      const float* __restrict__ W,
                      const float* __restrict__ Bias,
                      float* __restrict__ Out)
{
    __shared__ bf16_t As[2][MTILE * LDS_STRIDE];   // [m][k] bf16
    __shared__ bf16_t Bs[2][NTILE * LDS_STRIDE];   // [n][k] bf16 (transposed)

    const int tid  = threadIdx.x;
    const int lane = tid & 31;
    const int wave = tid >> 5;
    const int wm   = wave >> 2;          // 0..1  -> 64 rows each
    const int wn   = wave & 3;           // 0..3  -> 32 cols each
    const int l16  = lane & 15;
    const int half = lane >> 4;          // 0 or 1

    const int p     = blockIdx.z;
    const int mbase = blockIdx.y * MTILE;
    const int nbase = blockIdx.x * NTILE;

    const float* __restrict__ Xg = X + (size_t)mbase * ROWSTRIDE + (size_t)p * FIN;
    const float* __restrict__ Wg = W + (size_t)p * FIN * FOUT + nbase;

    // A loader coordinates: 128 rows x 32 k, float4 per thread per pass
    const int ar = tid >> 3;             // 0..31 : A row within a pass
    const int ac = (tid & 7) * 4;        // A col (float4)
    // B loader coordinates: each thread owns one column n and a 16-long k-run
    const int bn2 = tid & 127;           // n : 0..127  (consecutive lanes -> coalesced)
    const int bkh = tid >> 7;            // 0/1 : k half (k = bkh*16 .. +15)

    v4f   aReg[4];
    float bCol[16];

    auto g2r = [&](int k0) {             // global fp32 -> regs (coalesced)
        #pragma unroll
        for (int ps = 0; ps < 4; ++ps) {
            const int row = ps * 32 + ar;
            aReg[ps] = *(const v4f*)(Xg + (size_t)row * ROWSTRIDE + k0 + ac);
        }
        #pragma unroll
        for (int i = 0; i < 16; ++i)
            bCol[i] = Wg[(size_t)(k0 + bkh * 16 + i) * FOUT + bn2];
    };

    auto r2l = [&](int buf) {            // regs -> bf16 -> LDS (vectorized stores)
        #pragma unroll
        for (int ps = 0; ps < 4; ++ps) {
            const int row = ps * 32 + ar;
            v4bf a4 = { (bf16_t)aReg[ps].x, (bf16_t)aReg[ps].y,
                        (bf16_t)aReg[ps].z, (bf16_t)aReg[ps].w };
            *(v4bf*)&As[buf][row * LDS_STRIDE + ac] = a4;
        }
        v8bf blo, bhi;
        #pragma unroll
        for (int i = 0; i < 8; ++i) {
            blo[i] = (bf16_t)bCol[i];
            bhi[i] = (bf16_t)bCol[8 + i];
        }
        bf16_t* brow = &Bs[buf][bn2 * LDS_STRIDE + bkh * 16];
        *(v8bf*)(brow)     = blo;        // 16B-aligned: bn2*80 + bkh*32
        *(v8bf*)(brow + 8) = bhi;
    };

    v8f acc[4][2];
    #pragma unroll
    for (int mf = 0; mf < 4; ++mf)
        #pragma unroll
        for (int nf = 0; nf < 2; ++nf)
            #pragma unroll
            for (int r = 0; r < 8; ++r) acc[mf][nf][r] = 0.0f;

    g2r(0);
    r2l(0);
    int buf = 0;

    for (int kt = 0; kt < KSTEPS; ++kt) {
        if (kt + 1 < KSTEPS) g2r((kt + 1) * KTILE);   // prefetch next tile
        __syncthreads();                              // tile kt visible in LDS

        // ---- load WMMA fragments from LDS ----
        FragAB a[4], b[2];
        #pragma unroll
        for (int mf = 0; mf < 4; ++mf) {
            // 16-bit A 16x32 layout: lane half 0 -> K 0..7 / 16..23,
            //                        lane half 1 -> K 8..15 / 24..31
            const bf16_t* base = &As[buf][(wm * 64 + mf * 16 + l16) * LDS_STRIDE];
            a[mf].h[0] = *(const v8bf*)(base + half * 8);
            a[mf].h[1] = *(const v8bf*)(base + 16 + half * 8);
        }
        #pragma unroll
        for (int nf = 0; nf < 2; ++nf) {
            // 16-bit B 32x16 layout: lane half 0 -> K 0..7 / 16..23, lane half 1 -> K 8..15 / 24..31
            const bf16_t* base = &Bs[buf][(wn * 32 + nf * 16 + l16) * LDS_STRIDE];
            b[nf].h[0] = *(const v8bf*)(base + half * 8);
            b[nf].h[1] = *(const v8bf*)(base + 16 + half * 8);
        }

        // ---- 8 WMMAs per k-step per wave ----
        #pragma unroll
        for (int mf = 0; mf < 4; ++mf)
            #pragma unroll
            for (int nf = 0; nf < 2; ++nf)
                acc[mf][nf] = __builtin_amdgcn_wmma_f32_16x16x32_bf16(
                    /*neg_a=*/false, a[mf].v, /*neg_b=*/false, b[nf].v,
                    /*c_mod=*/(short)0, acc[mf][nf],
                    /*reuse_a=*/false, /*reuse_b=*/false);

        __syncthreads();                              // everyone done reading buf^1
        if (kt + 1 < KSTEPS) r2l(buf ^ 1);
        buf ^= 1;
    }

    // ---- epilogue: bias + store (C/D layout: lane=N, VGPR r = M row) ----
    #pragma unroll
    for (int mf = 0; mf < 4; ++mf) {
        const int m0 = mbase + wm * 64 + mf * 16 + half * 8;
        #pragma unroll
        for (int nf = 0; nf < 2; ++nf) {
            const int n = nbase + wn * 32 + nf * 16 + l16;
            const float bv = Bias[p * FOUT + n];
            float* o = Out + (size_t)m0 * OUTSTRIDE + (size_t)p * FOUT + n;
            #pragma unroll
            for (int r = 0; r < 8; ++r)
                o[(size_t)r * OUTSTRIDE] = acc[mf][nf][r] + bv;
        }
    }
}

extern "C" void kernel_launch(void* const* d_in, const int* in_sizes, int n_in,
                              void* d_out, int out_size, void* d_ws, size_t ws_size,
                              hipStream_t stream) {
    (void)in_sizes; (void)n_in; (void)out_size; (void)d_ws; (void)ws_size;
    const float* x      = (const float*)d_in[0];
    const float* kernel = (const float*)d_in[1];
    const float* bias   = (const float*)d_in[2];
    float* out          = (float*)d_out;

    dim3 grid(FOUT / NTILE, (B_ * S_) / MTILE, P_);   // (8, 64, 8)
    dim3 block(256);
    pdense_bf16_wmma<<<grid, block, 0, stream>>>(x, kernel, bias, out);
}
